// NTxent_Loss_v2_13013750907413
// MI455X (gfx1250) — matrix operand order repaired and checked by reading, not compiled, not run on hardware
//
#include <hip/hip_runtime.h>
#include <hip/hip_bf16.h>

typedef __attribute__((ext_vector_type(16))) __bf16 v16bf;
typedef __attribute__((ext_vector_type(8)))  __bf16 v8bf;
typedef __attribute__((ext_vector_type(8)))  float  v8f;

#define B_ROWS   4096
#define TWO_B    8192
#define DDIM     4096
#define KSTEPS   (DDIM / 32)
#define INV_T    10.0f     // 1/TEMP
#define NCT      64        // 8192/128 column tiles
#define NRT      32        // 4096/128 row tiles
#define A_STRIDE 40        // padded bf16 row stride (80B) to spread LDS banks
#define S_STRIDE 132       // padded f32 sim-tile stride

// ---------------- Kernel 1: exact f32 diagonal (positives) ----------------
__global__ __launch_bounds__(256) void diag_kernel(const float* __restrict__ im,
                                                   const float* __restrict__ s,
                                                   const float* __restrict__ sl,
                                                   float* __restrict__ diag) {
    int row  = blockIdx.x * 8 + (threadIdx.x >> 5);
    int lane = threadIdx.x & 31;
    const float* a = im + (size_t)row * DDIM;
    const float* b = s  + (size_t)row * DDIM;
    float acc = 0.f;
    for (int k = lane * 4; k < DDIM; k += 128) {
        float4 x = *(const float4*)(a + k);
        float4 y = *(const float4*)(b + k);
        acc += x.x * y.x + x.y * y.y + x.z * y.z + x.w * y.w;
    }
    for (int off = 16; off > 0; off >>= 1) acc += __shfl_xor(acc, off, 32);
    if (lane == 0)
        diag[row] = 0.5f * sl[(size_t)row * TWO_B + row] + 0.5f * acc;
}

// -------- staging helpers --------
__device__ __forceinline__ void load16(float* f, const float* __restrict__ g) {
    float4* f4 = (float4*)f;
#pragma unroll
    for (int q = 0; q < 4; ++q) f4[q] = *(const float4*)(g + q * 4);
}
__device__ __forceinline__ void cvtStore(__bf16* dst, const float* f) {
    v8bf lo, hi;
#pragma unroll
    for (int e = 0; e < 8; ++e) { lo[e] = (__bf16)f[e]; hi[e] = (__bf16)f[8 + e]; }
    *(v8bf*)&dst[0] = lo;
    *(v8bf*)&dst[8] = hi;
}

// ------- Kernel 2: pipelined bf16 WMMA GEMM + fused mask + softmax partials -------
__global__ __launch_bounds__(256) void gemm_partials(const float* __restrict__ im,
                                                     const float* __restrict__ s,
                                                     const float* __restrict__ sl,
                                                     const float* __restrict__ diag,
                                                     float2* __restrict__ rowPart,
                                                     float2* __restrict__ colPart) {
    const int colTile = blockIdx.x;          // 0..63
    const int rowTile = blockIdx.y;          // 0..31
    const int rowBase = rowTile * 128;
    const int colBase = colTile * 128;
    const int tid  = threadIdx.x;
    const int lane = tid & 31;
    const int wave = tid >> 5;
    const int waveM = (wave & 3) * 32;       // 4 waves along M
    const int waveN = (wave >> 2) * 64;      // 2 waves along N

    __shared__ __bf16 aT[2][128 * A_STRIDE];   // double-buffered A stage
    __shared__ __bf16 bT[2][128 * A_STRIDE];   // double-buffered B stage
    __shared__ float  simT[128 * S_STRIDE];

    v8f zero = {};
    v8f acc[2][4];
    for (int mi = 0; mi < 2; ++mi)
        for (int ni = 0; ni < 4; ++ni) acc[mi][ni] = zero;

    // staging: thread -> (row 0..127, 16-float half)
    const int lrow  = tid >> 1;
    const int lhalf = (tid & 1) * 16;
    const float* aG = im + (size_t)(rowBase + lrow) * DDIM + lhalf;
    const float* bG = s  + (size_t)(colBase + lrow) * DDIM + lhalf;
    const int stOff = lrow * A_STRIDE + lhalf;

    const int lr = lane & 15;   // 0..15
    const int hk = lane >> 4;   // half-wave k-group

    // ---- software pipeline prologue: stage K-block 0 ----
    float fa[16], fb[16];
    load16(fa, aG);
    load16(fb, bG);
    cvtStore(&aT[0][stOff], fa);
    cvtStore(&bT[0][stOff], fb);

    for (int i = 0; i < KSTEPS; ++i) {
        const bool more = (i + 1) < KSTEPS;
        // prefetch next K-block BEFORE the barrier: global latency overlaps
        // the barrier wait and the WMMA phase below
        if (more) {
            load16(fa, aG + (i + 1) * 32);
            load16(fb, bG + (i + 1) * 32);
        }
        __syncthreads();                        // stage (i&1) now fully written

        const __bf16* aC = aT[i & 1];
        const __bf16* bC = bT[i & 1];

        // A frag: row = waveM+mi*16+lr; elems 0..7 = K hk*8.., 8..15 = K 16+hk*8..
        v16bf afr[2], bfr[4];
#pragma unroll
        for (int mi = 0; mi < 2; ++mi) {
            int r = waveM + mi * 16 + lr;
            v8bf lo = *(const v8bf*)&aC[r * A_STRIDE + hk * 8];
            v8bf hi = *(const v8bf*)&aC[r * A_STRIDE + 16 + hk * 8];
            afr[mi] = __builtin_shufflevector(lo, hi, 0,1,2,3,4,5,6,7,8,9,10,11,12,13,14,15);
        }
        // B frag: col = waveN+ni*16+lr; elems 0..7 = K hk*8.., 8..15 = K 16+hk*8..
#pragma unroll
        for (int ni = 0; ni < 4; ++ni) {
            int n = waveN + ni * 16 + lr;
            v8bf lo = *(const v8bf*)&bC[n * A_STRIDE + hk * 8];
            v8bf hi = *(const v8bf*)&bC[n * A_STRIDE + 16 + hk * 8];
            bfr[ni] = __builtin_shufflevector(lo, hi, 0,1,2,3,4,5,6,7,8,9,10,11,12,13,14,15);
        }
#pragma unroll
        for (int mi = 0; mi < 2; ++mi)
#pragma unroll
            for (int ni = 0; ni < 4; ++ni)
                acc[mi][ni] = __builtin_amdgcn_wmma_f32_16x16x32_bf16(
                    false, afr[mi], false, bfr[ni], (short)0, acc[mi][ni], false, false);

        // write NEXT stage (disjoint buffer from the one being read this
        // iteration, so a single barrier per iteration suffices)
        if (more) {
            cvtStore(&aT[(i + 1) & 1][stOff], fa);
            cvtStore(&bT[(i + 1) & 1][stOff], fb);
        }
    }

    // spill C tiles: lane = col (l&15), VGPR r + 8*hk = row-in-tile
#pragma unroll
    for (int mi = 0; mi < 2; ++mi)
#pragma unroll
        for (int ni = 0; ni < 4; ++ni) {
            int col = waveN + ni * 16 + lr;
#pragma unroll
            for (int r = 0; r < 8; ++r) {
                int row = waveM + mi * 16 + hk * 8 + r;
                simT[row * S_STRIDE + col] = acc[mi][ni][r];
            }
        }
    __syncthreads();

    // ---- row pass: fuse sims_local, mask vs diag, online (max,sumexp); write x back
    const int rrow = tid >> 1;
    const int rseg = tid & 1;
    {
        int grow = rowBase + rrow;
        float dval = diag[grow];
        const float* slrow = sl + (size_t)grow * TWO_B + colBase;
        float m = -3.4e38f, ssum = 0.f;
        const bool secondHalf = (colBase >= B_ROWS);
#pragma unroll 4
        for (int c = 0; c < 64; ++c) {
            int col = rseg * 64 + c;
            float x = 0.5f * simT[rrow * S_STRIDE + col] + 0.5f * slrow[col];
            if (secondHalf && x > dval) x = 0.f;
            simT[rrow * S_STRIDE + col] = x;   // store fused value for column pass
            float lx = x * INV_T;
            if (lx > m) { ssum = ssum * __expf(m - lx) + 1.f; m = lx; }
            else          ssum += __expf(lx - m);
        }
        float m2 = __shfl_xor(m, 1, 32);
        float s2 = __shfl_xor(ssum, 1, 32);
        float M  = fmaxf(m, m2);
        float S  = ssum * __expf(m - M) + s2 * __expf(m2 - M);
        if (rseg == 0)
            rowPart[(size_t)grow * NCT + colTile] = make_float2(M, S);
    }
    __syncthreads();

    // ---- column pass (t2i): only for first-half tiles, values already fused in LDS
    if (colBase < B_ROWS) {
        int ccol = tid >> 1;
        int gcol = colBase + ccol;
        float m = -3.4e38f, ssum = 0.f;
#pragma unroll 4
        for (int r = 0; r < 64; ++r) {
            int row = rseg * 64 + r;
            float lx = simT[row * S_STRIDE + ccol] * INV_T;
            if (lx > m) { ssum = ssum * __expf(m - lx) + 1.f; m = lx; }
            else          ssum += __expf(lx - m);
        }
        float m2 = __shfl_xor(m, 1, 32);
        float s2 = __shfl_xor(ssum, 1, 32);
        float M  = fmaxf(m, m2);
        float S  = ssum * __expf(m - M) + s2 * __expf(m2 - M);
        if (rseg == 0)
            colPart[(size_t)gcol * NRT + rowTile] = make_float2(M, S);
    }
}

// ---------------- Kernel 3: merge partials -> scalar loss ----------------
__global__ __launch_bounds__(256) void finalize_kernel(const float2* __restrict__ rowPart,
                                                       const float2* __restrict__ colPart,
                                                       const float* __restrict__ diag,
                                                       float* __restrict__ out) {
    int tid = threadIdx.x;
    float local = 0.f;
    for (int i = tid; i < B_ROWS; i += 256) {
        float M = -3.4e38f, S = 0.f;
        for (int t = 0; t < NCT; ++t) {
            float2 p = rowPart[(size_t)i * NCT + t];
            if (p.x > M) { S = S * __expf(M - p.x) + p.y; M = p.x; }
            else           S += p.y * __expf(p.x - M);
        }
        float lseR = M + __logf(S);
        float Mc = -3.4e38f, Sc = 0.f;
        for (int t = 0; t < NRT; ++t) {
            float2 p = colPart[(size_t)i * NRT + t];
            if (p.x > Mc) { Sc = Sc * __expf(Mc - p.x) + p.y; Mc = p.x; }
            else            Sc += p.y * __expf(p.x - Mc);
        }
        float lseC = Mc + __logf(Sc);
        float d = diag[i] * INV_T;
        local += (lseR - d) + (lseC - d);
    }
    __shared__ float red[256];
    red[tid] = local;
    __syncthreads();
    for (int s = 128; s > 0; s >>= 1) {
        if (tid < s) red[tid] += red[tid + s];
        __syncthreads();
    }
    if (tid == 0) out[0] = red[0] * (1.0f / (float)B_ROWS);
}

extern "C" void kernel_launch(void* const* d_in, const int* in_sizes, int n_in,
                              void* d_out, int out_size, void* d_ws, size_t ws_size,
                              hipStream_t stream) {
    const float* im = (const float*)d_in[0];        // [4096, 4096]
    const float* s  = (const float*)d_in[1];        // [8192, 4096]
    const float* sl = (const float*)d_in[2];        // [4096, 8192]
    float* out = (float*)d_out;

    char* ws = (char*)d_ws;
    float*  diag    = (float*)ws;                                       // 16 KB
    float2* rowPart = (float2*)(ws + 16384);                            // 2 MB
    float2* colPart = (float2*)(ws + 16384 + (size_t)B_ROWS * NCT * 8); // 1 MB

    diag_kernel<<<B_ROWS / 8, 256, 0, stream>>>(im, s, sl, diag);
    gemm_partials<<<dim3(NCT, NRT), 256, 0, stream>>>(im, s, sl, diag, rowPart, colPart);
    finalize_kernel<<<1, 256, 0, stream>>>(rowPart, colPart, diag, out);
}
